// SelfAttention_31499290149101
// MI455X (gfx1250) — compile-verified, matrix operand order checked by reading, not run on hardware
//
#include <hip/hip_runtime.h>

constexpr int CB = 4;      // batch
constexpr int CS = 2048;   // seq len
constexpr int CE = 1024;   // embed
constexpr int CH = 16;     // heads
constexpr int CD = 64;     // head dim

typedef __attribute__((ext_vector_type(16))) __bf16 bf16x16;
typedef __attribute__((ext_vector_type(8)))  float  f32x8;

union FragU { bf16x16 v; uint4 q[2]; };
union Pack8 { uint4 q; __bf16 h[8]; };
union Pack4 { uint2 u; __bf16 h[4]; };

__device__ __forceinline__ f32x8 wmma_bf16(bf16x16 a, bf16x16 b, f32x8 c) {
    return __builtin_amdgcn_wmma_f32_16x16x32_bf16(false, a, false, b, (short)0, c,
                                                   false, false);
}

// A/B fragment loader for 16x16x32 bf16 WMMA.
// A (16x32, row-major src): lane holds row (lane&15); element e -> K = 8g + (e<8 ? e : e+8)
// => two contiguous 16B loads at K offsets (koff+8g) and (koff+8g+16).
// Identical pattern serves the B operand when the source is the column-major
// mate stored row-major (rows of W, rows of K, rows of V^T).
__device__ __forceinline__ bf16x16 load_frag(const __bf16* rowptr, int koff, int g) {
    FragU f;
    f.q[0] = *reinterpret_cast<const uint4*>(rowptr + koff + 8 * g);
    f.q[1] = *reinterpret_cast<const uint4*>(rowptr + koff + 8 * g + 16);
    return f.v;
}

// ---------------------------------------------------------------- weight cvt
__global__ void cvt_f32_bf16(const float* __restrict__ src, __bf16* __restrict__ dst, int n) {
    int i = blockIdx.x * 256 + threadIdx.x;
    if (i < n) dst[i] = (__bf16)src[i];
}

// ------------------------------------------------------- per-head projection
// One wave: 16 rows x 64 cols of X_h times W^T (64x64).  transposed==1 stores
// result as [B,H,D,S] (for V), else [B,H,S,D] (for Q,K).
__global__ void proj_kernel(const float* __restrict__ X, const __bf16* __restrict__ Wb,
                            __bf16* __restrict__ dst, int transposed) {
    __shared__ __bf16 lds[4][16 * 64];
    const int lane = threadIdx.x & 31;
    const int wv   = threadIdx.x >> 5;
    const int w    = blockIdx.x * 4 + wv;           // 0 .. 8191
    const int h    = w % CH;
    const int rb   = w / CH;                        // 0 .. B*S/16-1
    const int rowg = rb * 16;
    const int n    = rowg / CS;
    const int l0   = rowg % CS;
    const int g = lane >> 4, ln = lane & 15;
    __bf16* tile = lds[wv];

    // stage 16x64 f32 -> bf16 LDS tile (coalesced float4 loads)
    for (int i = lane; i < 256; i += 32) {
        int r = i >> 4, c4 = i & 15;
        float4 f = *reinterpret_cast<const float4*>(
            X + ((size_t)(n * CS + l0 + r)) * CE + h * CD + c4 * 4);
        Pack4 p;
        p.h[0] = (__bf16)f.x; p.h[1] = (__bf16)f.y;
        p.h[2] = (__bf16)f.z; p.h[3] = (__bf16)f.w;
        *reinterpret_cast<uint2*>(tile + r * 64 + c4 * 4) = p.u;
    }
    __syncthreads();

    bf16x16 a0 = load_frag(tile + ln * 64, 0,  g);
    bf16x16 a1 = load_frag(tile + ln * 64, 32, g);

    for (int nt = 0; nt < 4; ++nt) {
        const int d = nt * 16 + ln;                 // output column (B col = row of W)
        bf16x16 b0 = load_frag(Wb + (size_t)d * 64, 0,  g);
        bf16x16 b1 = load_frag(Wb + (size_t)d * 64, 32, g);
        f32x8 c = {};
        c = wmma_bf16(a0, b0, c);
        c = wmma_bf16(a1, b1, c);
        if (!transposed) {
            __bf16* base = dst + (((size_t)(n * CH + h)) * CS + l0 + 8 * g) * CD + d;
            for (int r = 0; r < 8; ++r) base[(size_t)r * CD] = (__bf16)c[r];
        } else {
            Pack8 p;
            for (int r = 0; r < 8; ++r) p.h[r] = (__bf16)c[r];
            *reinterpret_cast<uint4*>(
                dst + (((size_t)(n * CH + h)) * CD + d) * CS + l0 + 8 * g) = p.q;
        }
    }
}

// ---------------------------------------------------------- flash attention
// One wave per (n, h, 16-query tile); 32 keys per iteration, online softmax.
__global__ void attn_kernel(const __bf16* __restrict__ Qb, const __bf16* __restrict__ Kb,
                            const __bf16* __restrict__ Vt, const int* __restrict__ mask,
                            __bf16* __restrict__ attn) {
    __shared__ __bf16 ldsP[4][16 * 32];
    const int lane = threadIdx.x & 31;
    const int wv   = threadIdx.x >> 5;
    const int w    = blockIdx.x * 4 + wv;           // 0 .. 8191
    const int qt   = w % (CS / 16);
    const int h    = (w / (CS / 16)) % CH;
    const int n    = w / ((CS / 16) * CH);
    const int q0   = qt * 16;
    const int g = lane >> 4, ln = lane & 15;
    __bf16* P = ldsP[wv];

    const size_t hb = ((size_t)(n * CH + h)) * CS;
    const size_t vb = ((size_t)(n * CH + h)) * CD;
    const __bf16* qrow = Qb + (hb + q0 + ln) * CD;
    bf16x16 a0 = load_frag(qrow, 0,  g);
    bf16x16 a1 = load_frag(qrow, 32, g);

    float m[8], l[8];
    f32x8 acc[4];
    for (int r = 0; r < 8; ++r) { m[r] = -3.0e38f; l[r] = 0.f; }
    for (int nt = 0; nt < 4; ++nt) acc[nt] = (f32x8){0, 0, 0, 0, 0, 0, 0, 0};
    const float sc = 0.03125f;                      // 1/sqrt(EMBED) = 1/32

    for (int j = 0; j < CS; j += 32) {
        // prefetch next iteration's K/V tiles (lowers to global_prefetch_b8)
        if (j + 32 < CS) {
            __builtin_prefetch(Kb + (hb + j + 32 + ln) * CD, 0, 0);
            __builtin_prefetch(Kb + (hb + j + 48 + ln) * CD, 0, 0);
            for (int nt = 0; nt < 4; ++nt)
                __builtin_prefetch(Vt + (vb + nt * 16 + ln) * CS + j + 32, 0, 0);
        }
        // ---- scores: two 16x16 tiles, K-depth 64
        f32x8 st[2];
        for (int t = 0; t < 2; ++t) {
            const __bf16* krow = Kb + (hb + j + 16 * t + ln) * CD;
            bf16x16 b0 = load_frag(krow, 0,  g);
            bf16x16 b1 = load_frag(krow, 32, g);
            f32x8 c = {};
            c = wmma_bf16(a0, b0, c);
            c = wmma_bf16(a1, b1, c);
            st[t] = c;
        }
        // ---- mask + scale (lane column = key j+16t+ln; rows q0+8g+r)
        for (int t = 0; t < 2; ++t) {
            const int key = j + 16 * t + ln;
            const int* mp = mask + ((size_t)n * CS + q0 + 8 * g) * CS + key;
            for (int r = 0; r < 8; ++r) {
                int mk = mp[(size_t)r * CS];
                st[t][r] = mk ? st[t][r] * sc : -1.0e20f;
            }
        }
        // ---- online softmax (row reductions across 16-lane halves)
        float rm[8], rs[8];
        for (int r = 0; r < 8; ++r) rm[r] = fmaxf(st[0][r], st[1][r]);
        for (int off = 1; off < 16; off <<= 1)
            for (int r = 0; r < 8; ++r) rm[r] = fmaxf(rm[r], __shfl_xor(rm[r], off, 32));
        float mn[8], alpha[8];
        for (int r = 0; r < 8; ++r) {
            mn[r] = fmaxf(m[r], rm[r]);
            alpha[r] = __expf(m[r] - mn[r]);
            m[r] = mn[r];
        }
        f32x8 p0, p1;
        for (int r = 0; r < 8; ++r) {
            p0[r] = __expf(st[0][r] - mn[r]);
            p1[r] = __expf(st[1][r] - mn[r]);
            rs[r] = p0[r] + p1[r];
        }
        for (int off = 1; off < 16; off <<= 1)
            for (int r = 0; r < 8; ++r) rs[r] += __shfl_xor(rs[r], off, 32);
        for (int r = 0; r < 8; ++r) l[r] = l[r] * alpha[r] + rs[r];
        for (int nt = 0; nt < 4; ++nt)
            for (int r = 0; r < 8; ++r) acc[nt][r] *= alpha[r];

        // ---- re-layout P (C layout -> A fragment) through LDS
        __syncthreads();
        for (int r = 0; r < 8; ++r) {
            P[(8 * g + r) * 32 + ln]      = (__bf16)p0[r];
            P[(8 * g + r) * 32 + 16 + ln] = (__bf16)p1[r];
        }
        __syncthreads();
        bf16x16 pA = load_frag(P + ln * 32, 0, g);

        // ---- acc += P @ V  (V^T rows are contiguous along keys)
        for (int nt = 0; nt < 4; ++nt) {
            const __bf16* vrow = Vt + (vb + nt * 16 + ln) * CS + j;
            FragU f;
            f.q[0] = *reinterpret_cast<const uint4*>(vrow + 8 * g);
            f.q[1] = *reinterpret_cast<const uint4*>(vrow + 8 * g + 16);
            acc[nt] = wmma_bf16(pA, f.v, acc[nt]);
        }
    }

    for (int r = 0; r < 8; ++r) l[r] = 1.0f / l[r];
    for (int nt = 0; nt < 4; ++nt) {
        __bf16* base = attn + ((size_t)n * CS + q0 + 8 * g) * CE + h * CD + nt * 16 + ln;
        for (int r = 0; r < 8; ++r) base[(size_t)r * CE] = (__bf16)(acc[nt][r] * l[r]);
    }
}

// ------------------------------------------------------- output projection
// One wave computes a 16x64 f32 block of out = A @ Wo^T + bo, K = 1024.
// A fragment is loaded once per k-step and reused across 4 N-tiles (4x less
// L2 traffic, 4 independent back-to-back WMMAs per A load).
__global__ void outproj_kernel(const __bf16* __restrict__ A, const __bf16* __restrict__ Wb,
                               const float* __restrict__ bo, float* __restrict__ out) {
    const int lane = threadIdx.x & 31;
    const int wv   = threadIdx.x >> 5;
    const int w    = blockIdx.x * 4 + wv;           // 0 .. 8191
    const int cbk  = w % (CE / 64);                 // 16 column blocks of 64
    const int rb   = w / (CE / 64);                 // 512 row blocks of 16
    const int row0 = rb * 16, c0 = cbk * 64;
    const int g = lane >> 4, ln = lane & 15;

    const __bf16* arow = A + ((size_t)(row0 + ln)) * CE;
    f32x8 acc[4];
    for (int nt = 0; nt < 4; ++nt) acc[nt] = (f32x8){0, 0, 0, 0, 0, 0, 0, 0};

    for (int k0 = 0; k0 < CE; k0 += 32) {
        bf16x16 a = load_frag(arow, k0, g);
        for (int nt = 0; nt < 4; ++nt) {
            const __bf16* brow = Wb + ((size_t)(c0 + nt * 16 + ln)) * CE;
            bf16x16 b = load_frag(brow, k0, g);
            acc[nt] = wmma_bf16(a, b, acc[nt]);
        }
    }
    for (int nt = 0; nt < 4; ++nt) {
        const int col = c0 + nt * 16 + ln;
        const float bias = bo[col];
        float* obase = out + ((size_t)(row0 + 8 * g)) * CE + col;
        for (int r = 0; r < 8; ++r) obase[(size_t)r * CE] = acc[nt][r] + bias;
    }
}

// --------------------------------------------------------------------------
extern "C" void kernel_launch(void* const* d_in, const int* in_sizes, int n_in,
                              void* d_out, int out_size, void* d_ws, size_t ws_size,
                              hipStream_t stream) {
    const float* values = (const float*)d_in[0];
    const float* keys   = (const float*)d_in[1];
    const float* query  = (const float*)d_in[2];
    const int*   mask   = (const int*)d_in[3];
    const float* Wv     = (const float*)d_in[4];
    const float* Wk     = (const float*)d_in[5];
    const float* Wq     = (const float*)d_in[6];
    const float* Wo     = (const float*)d_in[7];
    const float* bo     = (const float*)d_in[8];
    float* out = (float*)d_out;

    char* ws = (char*)d_ws;
    __bf16* Wq_b = (__bf16*)(ws + 0);
    __bf16* Wk_b = (__bf16*)(ws + 8192);
    __bf16* Wv_b = (__bf16*)(ws + 16384);
    __bf16* Wo_b = (__bf16*)(ws + 32768);
    size_t off = 32768 + (size_t)CE * CE * 2;
    const size_t qkv = (size_t)CB * CH * CS * CD * 2;
    __bf16* Qb = (__bf16*)(ws + off); off += qkv;
    __bf16* Kb = (__bf16*)(ws + off); off += qkv;
    __bf16* Vt = (__bf16*)(ws + off); off += qkv;   // [B,H,D,S]
    __bf16* At = (__bf16*)(ws + off);               // [B,S,E] attn output

    cvt_f32_bf16<<<16,   256, 0, stream>>>(Wq, Wq_b, CD * CD);
    cvt_f32_bf16<<<16,   256, 0, stream>>>(Wk, Wk_b, CD * CD);
    cvt_f32_bf16<<<16,   256, 0, stream>>>(Wv, Wv_b, CD * CD);
    cvt_f32_bf16<<<4096, 256, 0, stream>>>(Wo, Wo_b, CE * CE);

    proj_kernel<<<2048, 128, 0, stream>>>(query,  Wq_b, Qb, 0);
    proj_kernel<<<2048, 128, 0, stream>>>(keys,   Wk_b, Kb, 0);
    proj_kernel<<<2048, 128, 0, stream>>>(values, Wv_b, Vt, 1);

    attn_kernel<<<2048, 128, 0, stream>>>(Qb, Kb, Vt, mask, At);

    outproj_kernel<<<2048, 128, 0, stream>>>(At, Wo_b, bo, out);
}
